// Encoder_57861799412266
// MI455X (gfx1250) — matrix operand order repaired and checked by reading, not compiled, not run on hardware
//
#include <hip/hip_runtime.h>

#define T_ 256
#define B_ 32
#define E_ 512
#define U_ 1024
#define G_ 3072            // 3*U
#define RBLK_PER_DIR 24    // 24 blocks * 8 waves = 192 N-tiles (each does full M=32)
#define RTHREADS 256
#define LDS_COLSTRIDE 1032 // 1024 + 8 bf16 pad -> avoids 16-way bank conflicts
#define LDS_PER_WAVE (16 * LDS_COLSTRIDE)          // bf16 elements
#define LDS_BYTES (8 * LDS_PER_WAVE * 2)           // 264192 B < 320KB WGP LDS

typedef __attribute__((ext_vector_type(16))) __bf16 v16bf;
typedef __attribute__((ext_vector_type(8)))  __bf16 v8bf;
typedef __attribute__((ext_vector_type(8)))  float  v8f;

// ---- A-operand load matching ISA 16-bit A 16x32 layout:
// lane<16: K = [kb..kb+7] then [16+kb..16+kb+7] with kb=0 ; lane>=16: kb=8
__device__ __forceinline__ v16bf load_a16(const __bf16* row, int k0, int kb) {
  v8bf lo = *(const v8bf*)(row + k0 + kb);
  v8bf hi = *(const v8bf*)(row + k0 + 16 + kb);
  return __builtin_shufflevector(lo, hi, 0,1,2,3,4,5,6,7,8,9,10,11,12,13,14,15);
}

// ---- B-operand: K = [off..off+7] then [off+16..off+23] of one column (two 16-byte loads)
__device__ __forceinline__ v16bf load_b16(const __bf16* col, int off) {
  v8bf lo = *(const v8bf*)(col + off);
  v8bf hi = *(const v8bf*)(col + off + 16);
  return __builtin_shufflevector(lo, hi, 0,1,2,3,4,5,6,7,8,9,10,11,12,13,14,15);
}

__device__ __forceinline__ v8f wmma_bf16(v16bf a, v16bf b, v8f c) {
  // 8 args: (neg_a, A, neg_b, B, c_mod, C, reuse_a, reuse_b)
  return __builtin_amdgcn_wmma_f32_16x16x32_bf16(false, a, false, b, (short)0, c, false, false);
}

// ---------------- prep kernels ----------------

// dst[n*K + k] = (bf16) src[k*N + n]
__global__ void transpose_bf16_kernel(const float* __restrict__ src,
                                      __bf16* __restrict__ dst, int K, int N) {
  long idx = (long)blockIdx.x * blockDim.x + threadIdx.x;
  if (idx >= (long)K * N) return;
  int  n = (int)(idx % N);
  long k = idx / N;
  dst[(long)n * K + k] = (__bf16)src[idx];
}

// xe[(t*B+b)*E + e] = bf16(emb[x[b][t]][e])
__global__ void gather_embed_kernel(const int* __restrict__ x,
                                    const float* __restrict__ emb,
                                    __bf16* __restrict__ xe) {
  long idx = (long)blockIdx.x * blockDim.x + threadIdx.x;
  if (idx >= (long)T_ * B_ * E_) return;
  int  e = (int)(idx & (E_ - 1));
  long r = idx >> 9;          // / E_
  int  t = (int)(r >> 5);     // / B_
  int  b = (int)(r & 31);
  int  tok = x[b * T_ + t];
  xe[idx] = (__bf16)emb[(long)tok * E_ + e];
}

__global__ void init_state_kernel(const float* __restrict__ h0f,
                                  const float* __restrict__ h0b,
                                  __bf16* __restrict__ h_bf,
                                  float* __restrict__ hbuf,
                                  unsigned* __restrict__ bar) {
  int idx = blockIdx.x * blockDim.x + threadIdx.x;
  if (idx < B_ * U_) {
    float vf = h0f[idx], vb = h0b[idx];
    hbuf[idx]            = vf;
    hbuf[B_ * U_ + idx]  = vb;
    h_bf[idx]            = (__bf16)vf;
    h_bf[B_ * U_ + idx]  = (__bf16)vb;
  }
  if (idx == 0) { bar[0] = 0u; bar[1] = 0u; }
}

// ---------------- big input-projection GEMM: xW = xe @ W + b0 ----------------
// Each wave: 4 M-tiles (64 rows) x 1 N-tile, B reused across the 4 WMMAs.
__global__ __launch_bounds__(256) void xw_gemm_kernel(const __bf16* __restrict__ xe,
                                                      const __bf16* __restrict__ Wt,
                                                      const float* __restrict__ b_f,
                                                      const float* __restrict__ b_b,
                                                      float* __restrict__ xW) {
  const int lane = threadIdx.x & 31;
  const int wave = threadIdx.x >> 5;
  const long NT = G_ / 16;               // 192
  const long MG = (long)(T_ * B_) / 64;  // 128 groups of 64 rows
  long job = (long)blockIdx.x * 8 + wave;
  int  dir = (int)(job / (MG * NT));
  long rem = job % (MG * NT);
  int  mg = (int)(rem / NT);
  int  nt = (int)(rem % NT);

  const int m   = lane & 15;
  const int kb  = (lane >> 4) * 8;
  const int kb2 = (lane >> 4) * 16;
  const int n   = nt * 16 + m;

  const __bf16* A0   = xe + (long)(mg * 64 + m) * E_;
  const __bf16* Bcol = Wt + (long)dir * G_ * E_ + (long)n * E_;

  v8f acc[4];
  #pragma unroll
  for (int i = 0; i < 4; ++i) acc[i] = (v8f){0.f,0.f,0.f,0.f,0.f,0.f,0.f,0.f};

  for (int k0 = 0; k0 < E_; k0 += 32) {
    v16bf b = load_b16(Bcol, k0 + kb);
    #pragma unroll
    for (int i = 0; i < 4; ++i) {
      v16bf a = load_a16(A0 + (long)i * 16 * E_, k0, kb);
      acc[i] = wmma_bf16(a, b, acc[i]);
    }
  }

  const float bv = (dir ? b_b : b_f)[n];   // b[0] row
  float* dst = xW + (long)dir * (T_ * B_) * (long)G_;
  const int rowBase = (lane >> 4) * 8;
  #pragma unroll
  for (int i = 0; i < 4; ++i)
    #pragma unroll
    for (int j = 0; j < 8; ++j)
      dst[(long)(mg * 64 + i * 16 + rowBase + j) * G_ + n] = acc[i][j] + bv;
}

// ---------------- persistent recurrence kernel ----------------

__device__ __forceinline__ void grid_sync(unsigned* cnt, unsigned* gen,
                                          unsigned nblk, unsigned& mygen) {
  __syncthreads();
  if (threadIdx.x == 0) {
    __threadfence();
    unsigned prev = __hip_atomic_fetch_add(cnt, 1u, __ATOMIC_ACQ_REL, __HIP_MEMORY_SCOPE_AGENT);
    if (prev == nblk - 1) {
      __hip_atomic_store(cnt, 0u, __ATOMIC_RELAXED, __HIP_MEMORY_SCOPE_AGENT);
      __hip_atomic_fetch_add(gen, 1u, __ATOMIC_RELEASE, __HIP_MEMORY_SCOPE_AGENT);
    } else {
      while (__hip_atomic_load(gen, __ATOMIC_ACQUIRE, __HIP_MEMORY_SCOPE_AGENT) <= mygen) {
        __builtin_amdgcn_s_sleep(2);
      }
    }
  }
  mygen++;
  __syncthreads();
}

__global__ __launch_bounds__(RTHREADS) void gru_rec_kernel(
    const float* __restrict__ xW,    // [2][T][B][G]
    const __bf16* __restrict__ Ut,   // [2][G][U]  (transposed, bf16)
    const float* __restrict__ b_f,   // [2][G]
    const float* __restrict__ b_b,
    __bf16* h_bf,                    // [2][B][U] bf16 state (A-operand)
    float*  hbuf,                    // [2][B][U] f32 state
    float*  hU,                      // [2][B][G] per-step GEMM output
    float* __restrict__ out,         // y [B][T][2U] ++ h_f [B][U] ++ h_b [B][U]
    unsigned* bar_cnt, unsigned* bar_gen) {
  extern __shared__ __bf16 ldsB[];                 // [8 waves][16 cols][LDS_COLSTRIDE]
  const int lane = threadIdx.x & 31;
  const int wave = threadIdx.x >> 5;
  const int dir  = blockIdx.x / RBLK_PER_DIR;
  const int blk  = blockIdx.x % RBLK_PER_DIR;
  const int nt   = blk * 8 + wave;       // 0..191 : N-tile; wave covers full M=32
  const int m    = lane & 15;
  const int kb   = (lane >> 4) * 8;
  const int kb2  = (lane >> 4) * 16;
  const int n    = nt * 16 + m;

  __bf16* ldsW = ldsB + wave * LDS_PER_WAVE;

  // ---- stage this wave's B slice (16 cols x 1024 K, bf16) into LDS once ----
  {
    const __bf16* src = Ut + (long)dir * G_ * U_ + (long)(nt * 16) * U_;
    for (int i = lane * 8; i < 16 * 1024; i += 32 * 8) {
      int col = i >> 10;
      int k   = i & 1023;
      *(v8bf*)(ldsW + col * LDS_COLSTRIDE + k) = *(const v8bf*)(src + (long)col * U_ + k);
    }
  }
  __syncthreads();

  const __bf16* ldsCol = ldsW + m * LDS_COLSTRIDE;   // lane's column in LDS
  const float   b1v  = ((dir ? b_b : b_f) + G_)[n];  // b[1] row
  __bf16* hA  = h_bf + dir * B_ * U_;
  float*  hUo = hU   + dir * B_ * G_;
  float*  hB  = hbuf + dir * B_ * U_;
  const int rowBase = (lane >> 4) * 8;

  unsigned mygen = 0;
  const unsigned nblk = 2 * RBLK_PER_DIR;
  const int gThread = blk * RTHREADS + threadIdx.x;
  const int nThreadsDir = RBLK_PER_DIR * RTHREADS;

  const __bf16* ArowLo = hA + (long)m * U_;          // M rows 0..15
  const __bf16* ArowHi = hA + (long)(16 + m) * U_;   // M rows 16..31

  for (int s = 0; s < T_; ++s) {
    // ---- phase 1: hU = h @ U_r + b1   (M=32, N=3072, K=1024) ----
    v8f acc0 = {0.f,0.f,0.f,0.f,0.f,0.f,0.f,0.f};
    v8f acc1 = {0.f,0.f,0.f,0.f,0.f,0.f,0.f,0.f};
    #pragma unroll 4
    for (int k0 = 0; k0 < U_; k0 += 32) {
      v16bf b  = load_b16(ldsCol, k0 + kb);          // LDS-resident B, reused 2x
      v16bf a0 = load_a16(ArowLo, k0, kb);
      v16bf a1 = load_a16(ArowHi, k0, kb);
      acc0 = wmma_bf16(a0, b, acc0);
      acc1 = wmma_bf16(a1, b, acc1);
    }
    #pragma unroll
    for (int j = 0; j < 8; ++j) {
      hUo[(long)(rowBase + j) * G_ + n]      = acc0[j] + b1v;
      hUo[(long)(16 + rowBase + j) * G_ + n] = acc1[j] + b1v;
    }

    grid_sync(bar_cnt, bar_gen, nblk, mygen);

    // ---- phase 2: gates + state update ----
    const int t = dir ? (T_ - 1 - s) : s;
    const float* xWt = xW + ((long)dir * T_ + t) * (long)(B_ * G_);
    for (int idx = gThread; idx < B_ * U_; idx += nThreadsDir) {
      int bb = idx >> 10;           // / U_
      int u  = idx & (U_ - 1);
      const float* xrow = xWt + (long)bb * G_;
      const float* hrow = hUo + (long)bb * G_;
      float xz = xrow[u], xr = xrow[u + U_], xh = xrow[u + 2 * U_];
      float hz = hrow[u], hr = hrow[u + U_], hn = hrow[u + 2 * U_];
      float z  = 1.f / (1.f + __expf(-(xz + hz)));
      float r  = 1.f / (1.f + __expf(-(xr + hr)));
      float nn = tanhf(xh + r * hn);
      float hp = hB[idx];
      float hnew = z * hp + (1.f - z) * nn;
      hB[idx] = hnew;
      hA[idx] = (__bf16)hnew;
      out[((long)bb * T_ + t) * (2 * U_) + dir * U_ + u] = hnew;
      if (s == T_ - 1)
        out[(long)B_ * T_ * 2 * U_ + (long)dir * B_ * U_ + idx] = hnew;
    }

    grid_sync(bar_cnt, bar_gen, nblk, mygen);
  }
}

// ---------------- host launcher ----------------

extern "C" void kernel_launch(void* const* d_in, const int* in_sizes, int n_in,
                              void* d_out, int out_size, void* d_ws, size_t ws_size,
                              hipStream_t stream) {
  (void)in_sizes; (void)n_in; (void)out_size; (void)ws_size;
  const int*   x    = (const int*)  d_in[0];
  const float* emb  = (const float*)d_in[1];
  const float* W_f  = (const float*)d_in[2];
  const float* Uf   = (const float*)d_in[3];
  const float* b_f  = (const float*)d_in[4];
  const float* W_b  = (const float*)d_in[5];
  const float* Ub   = (const float*)d_in[6];
  const float* b_b  = (const float*)d_in[7];
  const float* h0f  = (const float*)d_in[8];
  const float* h0b  = (const float*)d_in[9];
  float* out = (float*)d_out;

  char* p = (char*)d_ws;
  auto alloc = [&](size_t bytes) -> char* {
    char* r = p; p += (bytes + 255) & ~(size_t)255; return r;
  };
  __bf16*   xe   = (__bf16*)  alloc((size_t)T_ * B_ * E_ * 2);
  __bf16*   Wt   = (__bf16*)  alloc((size_t)2 * G_ * E_ * 2);
  __bf16*   Ut   = (__bf16*)  alloc((size_t)2 * G_ * U_ * 2);
  __bf16*   h_bf = (__bf16*)  alloc((size_t)2 * B_ * U_ * 2);
  float*    hbuf = (float*)   alloc((size_t)2 * B_ * U_ * 4);
  float*    hU   = (float*)   alloc((size_t)2 * B_ * G_ * 4);
  unsigned* bar  = (unsigned*)alloc(256);
  float*    xW   = (float*)   alloc((size_t)2 * T_ * B_ * G_ * 4);

  // weight transposes + bf16 conversion
  transpose_bf16_kernel<<<(E_ * G_ + 255) / 256, 256, 0, stream>>>(W_f, Wt, E_, G_);
  transpose_bf16_kernel<<<(E_ * G_ + 255) / 256, 256, 0, stream>>>(W_b, Wt + (size_t)G_ * E_, E_, G_);
  transpose_bf16_kernel<<<(U_ * G_ + 255) / 256, 256, 0, stream>>>(Uf, Ut, U_, G_);
  transpose_bf16_kernel<<<(U_ * G_ + 255) / 256, 256, 0, stream>>>(Ub, Ut + (size_t)G_ * U_, U_, G_);

  // embedding gather
  gather_embed_kernel<<<(int)(((size_t)T_ * B_ * E_ + 255) / 256), 256, 0, stream>>>(x, emb, xe);

  // state + barrier init
  init_state_kernel<<<(B_ * U_ + 255) / 256, 256, 0, stream>>>(h0f, h0b, h_bf, hbuf, bar);

  // big GEMM: 2 dirs * 128 row-groups(64) * 192 Ntiles, 8 wave-jobs per block
  xw_gemm_kernel<<<(2 * 128 * 192) / 8, 256, 0, stream>>>(xe, Wt, b_f, b_b, xW);

  // persistent bidirectional recurrence (264 KB dynamic LDS per block)
  gru_rec_kernel<<<2 * RBLK_PER_DIR, RTHREADS, LDS_BYTES, stream>>>(
      xW, Ut, b_f, b_b, h_bf, hbuf, hU, out, bar, bar + 1);
}
